// MultiHeadAttention2_88734024335589
// MI455X (gfx1250) — matrix operand order repaired and checked by reading, not compiled, not run on hardware
//
#include <hip/hip_runtime.h>
#include <hip/hip_bf16.h>

typedef __bf16 bf16_t;
typedef __attribute__((ext_vector_type(16))) __bf16 v16bf;
typedef __attribute__((ext_vector_type(8)))  float  v8f;

#define Bm  8
#define Sm  1024
#define Dm  1024
#define Hn  16
#define DKm 64

#if defined(__has_builtin)
#if __has_builtin(__builtin_amdgcn_global_load_async_to_lds_b128)
#define ASYNC_OK 1
#endif
#endif

// builtin expects: (vector_size(16) int AS1*, vector_size(16) int AS3*, Ii, Ii)
typedef int v4i_vs __attribute__((vector_size(16)));
typedef __attribute__((address_space(1))) v4i_vs gv4i;
typedef __attribute__((address_space(3))) v4i_vs lv4i;

// 16-byte global -> LDS copy; async path when the toolchain has it.
__device__ __forceinline__ void async_copy_b128(const void* g, void* l) {
#if defined(ASYNC_OK)
    __builtin_amdgcn_global_load_async_to_lds_b128((gv4i*)g, (lv4i*)l, 0, 0);
#else
    *(uint4*)l = *(const uint4*)g;
#endif
}
__device__ __forceinline__ void async_wait() {
#if defined(ASYNC_OK)
    asm volatile("s_wait_asynccnt 0x0" ::: "memory");
#endif
}

union FragU { v16bf v; uint4 q[2]; };

// D = A*B + C ; A 16x32 bf16, B 32x16 bf16, C/D 16x16 f32
__device__ __forceinline__ v8f wmma_bf16(v16bf a, v16bf b, v8f c) {
    return __builtin_amdgcn_wmma_f32_16x16x32_bf16(false, a, false, b,
                                                   (short)0, c, false, false);
}

// A fragment: lane -> M row (lane&15 + row0), regs -> K.
__device__ __forceinline__ v16bf frag_a(const bf16_t* p, int row0, int stride,
                                        int kbase, int lane) {
    int r = row0 + (lane & 15);
    int k = kbase + ((lane & 16) ? 8 : 0);
    FragU f;
    f.q[0] = *(const uint4*)(p + (size_t)r * stride + k);
    f.q[1] = *(const uint4*)(p + (size_t)r * stride + k + 16);
    return f.v;
}

// B fragment: lane -> N col (n0 + lane&15), regs -> K (same order as frag_a).
// tile stored [K][N] with row stride `stride`.
__device__ __forceinline__ v16bf frag_b(const bf16_t* p, int kbase, int stride,
                                        int n0, int lane) {
    int n = n0 + (lane & 15);
    int k = kbase + ((lane & 16) ? 8 : 0);
    v16bf v;
    #pragma unroll
    for (int s = 0; s < 8; ++s) {
        v[s]     = p[(size_t)(k + s) * stride + n];
        v[s + 8] = p[(size_t)(k + 16 + s) * stride + n];
    }
    return v;
}

// ---------------------------------------------------------------------------
// Kernel 1: X[8192,1024] @ W.T + bias, optional RoPE over paired columns
// (n, n+512), bf16 output in head layout [b][h][s][dk].
// grid = (64 M-tiles of 128, 16 column-pair tiles), block = 128 (4 waves).
// Each wave owns two 16-row M strips -> 8 WMMAs per K-step.
// ---------------------------------------------------------------------------
__global__ __launch_bounds__(128)
void proj_rope(const float* __restrict__ X, const float* __restrict__ W,
               const float* __restrict__ bias, const float* __restrict__ cosT,
               const float* __restrict__ sinT, bf16_t* __restrict__ Out,
               int do_rope) {
    __shared__ __align__(16) bf16_t As[128 * 32];      // [M][K]
    __shared__ __align__(16) bf16_t Bs[2][32 * 32];    // [K][N] (transposed W)
    const int tid  = threadIdx.x;
    const int lane = tid & 31;
    const int wv   = tid >> 5;
    const int m0   = blockIdx.x * 128;
    const int n0   = blockIdx.y * 32;                  // low group; high = +512

    v8f acc[2][2][2] = {};                             // [strip][lo/hi][nfrag]

    for (int kt = 0; kt < Dm; kt += 32) {
        // stage A: 128x32 f32 -> bf16
        for (int i = tid; i < 128 * 32 / 4; i += 128) {
            int e = i * 4, row = e >> 5, col = e & 31;
            float4 f = *(const float4*)(X + (size_t)(m0 + row) * Dm + kt + col);
            bf16_t* ap = &As[row * 32 + col];
            ap[0] = (bf16_t)f.x; ap[1] = (bf16_t)f.y;
            ap[2] = (bf16_t)f.z; ap[3] = (bf16_t)f.w;
        }
        // stage B transposed: Bs[g][k][n] = W[n0+(g?512:0)+n][kt+k]
        for (int i = tid; i < 2 * 32 * 32 / 4; i += 128) {
            int g = i >> 8, ii = i & 255;
            int e = ii * 4, n = e >> 5, c = e & 31;
            int wrow = n0 + (g ? 512 : 0) + n;
            float4 f = *(const float4*)(W + (size_t)wrow * Dm + kt + c);
            bf16_t* bp = Bs[g];
            bp[(c + 0) * 32 + n] = (bf16_t)f.x;
            bp[(c + 1) * 32 + n] = (bf16_t)f.y;
            bp[(c + 2) * 32 + n] = (bf16_t)f.z;
            bp[(c + 3) * 32 + n] = (bf16_t)f.w;
        }
        __syncthreads();
        v16bf a0 = frag_a(As, wv * 16, 32, 0, lane);
        v16bf a1 = frag_a(As, 64 + wv * 16, 32, 0, lane);
        #pragma unroll
        for (int g = 0; g < 2; ++g)
            #pragma unroll
            for (int f = 0; f < 2; ++f) {
                v16bf bb = frag_b(Bs[g], 0, 32, f * 16, lane);
                acc[0][g][f] = wmma_bf16(a0, bb, acc[0][g][f]);
                acc[1][g][f] = wmma_bf16(a1, bb, acc[1][g][f]);
            }
        __syncthreads();
    }

    const int mloc = (lane & 16) ? 8 : 0;
    const int ncol = lane & 15;
    #pragma unroll
    for (int st = 0; st < 2; ++st)
        #pragma unroll
        for (int f = 0; f < 2; ++f) {
            int nLo = n0 + f * 16 + ncol;
            int nHi = nLo + 512;
            float bL = bias[nLo], bH = bias[nHi];
            #pragma unroll
            for (int r = 0; r < 8; ++r) {
                int mg = m0 + st * 64 + wv * 16 + mloc + r;
                int s  = mg & (Sm - 1);
                int bi = mg >> 10;
                float xl = acc[st][0][f][r] + bL;
                float xh = acc[st][1][f][r] + bH;
                float yl = xl, yh = xh;
                if (do_rope) {
                    float cl = cosT[(size_t)s * Dm + nLo], sl = sinT[(size_t)s * Dm + nLo];
                    float ch = cosT[(size_t)s * Dm + nHi], sh = sinT[(size_t)s * Dm + nHi];
                    yl = xl * cl - xh * sl;   // rot_half low  = -x_high
                    yh = xh * ch + xl * sh;   // rot_half high = +x_low
                }
                Out[(((size_t)bi * Hn + (nLo >> 6)) * Sm + s) * DKm + (nLo & 63)] = (bf16_t)yl;
                Out[(((size_t)bi * Hn + (nHi >> 6)) * Sm + s) * DKm + (nHi & 63)] = (bf16_t)yh;
            }
        }
}

// ---------------------------------------------------------------------------
// Kernel 2: flash attention. grid = (S/64, H, B), block = 128 (4 waves).
// Each wave owns 16 query rows; block streams 32-wide KV tiles via LDS.
// V tile is staged with async global->LDS copies; next tiles prefetched.
// ---------------------------------------------------------------------------
__global__ __launch_bounds__(128)
void flash_attn(const bf16_t* __restrict__ qh, const bf16_t* __restrict__ kh,
                const bf16_t* __restrict__ vh, const float* __restrict__ mask,
                const float* __restrict__ shortM, const float* __restrict__ pbias,
                bf16_t* __restrict__ concatO) {
    __shared__ __align__(16) bf16_t KtT[64 * 32];      // [dk][kv]  (B of Q*K^T)
    __shared__ __align__(16) bf16_t Vt [32 * 64];      // [kv][dk]  (B of P*V)
    __shared__ __align__(16) bf16_t Pb [4][16 * 32];   // per-wave P re-layout
    const int tid = threadIdx.x, lane = tid & 31, wv = tid >> 5;
    const int b = blockIdx.z, h = blockIdx.y;
    const int q0 = blockIdx.x * 64 + wv * 16;
    const size_t headBase = ((size_t)b * Hn + h) * Sm * DKm;
    const float sig = 1.f / (1.f + __expf(-pbias[h]));

    // preload Q A-fragments (K = dk in two 32-steps)
    v16bf Aq0 = frag_a(qh + headBase, q0, DKm, 0,  lane);
    v16bf Aq1 = frag_a(qh + headBase, q0, DKm, 32, lane);

    v8f Oacc[4] = {};
    float mrow[8], lrow[8];
    #pragma unroll
    for (int r = 0; r < 8; ++r) { mrow[r] = -1e30f; lrow[r] = 0.f; }
    const int mloc = (lane & 16) ? 8 : 0;
    const int ncol = lane & 15;

    for (int j = 0; j < Sm; j += 32) {
        // async stage V tile straight: Vt[kv][dk]
        for (int i = tid; i < 32 * 64 / 8; i += 128) {
            int e = i * 8, kv = e >> 6, dk = e & 63;
            async_copy_b128(vh + headBase + (size_t)(j + kv) * DKm + dk,
                            Vt + kv * 64 + dk);
        }
        // stage K tile transposed: KtT[dk][kv] = kh[j+kv][dk]
        for (int i = tid; i < 32 * 64 / 8; i += 128) {
            int e = i * 8, kv = e >> 6, dk = e & 63;
            const bf16_t* src = kh + headBase + (size_t)(j + kv) * DKm + dk;
            #pragma unroll
            for (int t = 0; t < 8; ++t) KtT[(dk + t) * 32 + kv] = src[t];
        }
        // prefetch next KV tiles while this one computes
        if (j + 32 < Sm) {
            __builtin_prefetch(kh + headBase + (size_t)(j + 32 + lane) * DKm, 0, 1);
            __builtin_prefetch(vh + headBase + (size_t)(j + 32 + lane) * DKm, 0, 1);
        }
        async_wait();
        __syncthreads();

        // scores: 16 x 32 tile, contraction over dk = 64
        v8f sc[2] = {};
        #pragma unroll
        for (int f = 0; f < 2; ++f) {
            sc[f] = wmma_bf16(Aq0, frag_b(KtT, 0,  32, f * 16, lane), sc[f]);
            sc[f] = wmma_bf16(Aq1, frag_b(KtT, 32, 32, f * 16, lane), sc[f]);
        }

        // bias + online softmax
        float p[2][8];
        float mk0 = mask[b * Sm + j + ncol];
        float mk1 = mask[b * Sm + j + 16 + ncol];
        #pragma unroll
        for (int r = 0; r < 8; ++r) {
            int m = q0 + mloc + r;
            p[0][r] = sc[0][r] * 0.125f + mk0 + sig * shortM[(size_t)m * Sm + j + ncol];
            p[1][r] = sc[1][r] * 0.125f + mk1 + sig * shortM[(size_t)m * Sm + j + 16 + ncol];
        }
        float scale[8];
        #pragma unroll
        for (int r = 0; r < 8; ++r) {
            float vm = fmaxf(p[0][r], p[1][r]);
            #pragma unroll
            for (int off = 8; off >= 1; off >>= 1)
                vm = fmaxf(vm, __shfl_xor(vm, off, 16));
            float mn = fmaxf(mrow[r], vm);
            scale[r] = __expf(mrow[r] - mn);
            mrow[r]  = mn;
            float e0 = __expf(p[0][r] - mn), e1 = __expf(p[1][r] - mn);
            p[0][r] = e0; p[1][r] = e1;
            float rs = e0 + e1;
            #pragma unroll
            for (int off = 8; off >= 1; off >>= 1)
                rs += __shfl_xor(rs, off, 16);
            lrow[r] = lrow[r] * scale[r] + rs;
        }
        #pragma unroll
        for (int g = 0; g < 4; ++g)
            #pragma unroll
            for (int r = 0; r < 8; ++r) Oacc[g][r] *= scale[r];

        // C-layout P -> LDS -> A-layout fragment
        bf16_t* Pw = Pb[wv];
        #pragma unroll
        for (int f = 0; f < 2; ++f)
            #pragma unroll
            for (int r = 0; r < 8; ++r)
                Pw[(mloc + r) * 32 + f * 16 + ncol] = (bf16_t)p[f][r];
        __syncthreads();
        v16bf ap = frag_a(Pw, 0, 32, 0, lane);
        #pragma unroll
        for (int g = 0; g < 4; ++g)
            Oacc[g] = wmma_bf16(ap, frag_b(Vt, 0, 64, g * 16, lane), Oacc[g]);
        __syncthreads();
    }

    // epilogue: normalize, write concat layout [b][s][h*64+dk] as bf16
    #pragma unroll
    for (int g = 0; g < 4; ++g)
        #pragma unroll
        for (int r = 0; r < 8; ++r) {
            int m = q0 + mloc + r;
            float v = Oacc[g][r] / lrow[r];
            concatO[((size_t)b * Sm + m) * Dm + h * DKm + g * 16 + ncol] = (bf16_t)v;
        }
}

// ---------------------------------------------------------------------------
// Kernel 3: out = concat(bf16) @ Wo.T + bo, fp32 output.
// grid = (64 M-tiles of 128, 32 N-tiles), block = 128; async A-tile staging.
// ---------------------------------------------------------------------------
__global__ __launch_bounds__(128)
void out_proj(const bf16_t* __restrict__ Xb, const float* __restrict__ W,
              const float* __restrict__ bias, float* __restrict__ Out) {
    __shared__ __align__(16) bf16_t As[128 * 32];
    __shared__ __align__(16) bf16_t Bs[32 * 32];       // [K][N]
    const int tid = threadIdx.x, lane = tid & 31, wv = tid >> 5;
    const int m0 = blockIdx.x * 128, n0 = blockIdx.y * 32;
    v8f acc[2][2] = {};                                // [strip][nfrag]

    for (int kt = 0; kt < Dm; kt += 32) {
        for (int i = tid; i < 128 * 32 / 8; i += 128) {
            int e = i * 8, row = e >> 5, col = e & 31;
            async_copy_b128(Xb + (size_t)(m0 + row) * Dm + kt + col,
                            As + row * 32 + col);
        }
        for (int i = tid; i < 32 * 32 / 4; i += 128) {
            int e = i * 4, n = e >> 5, c = e & 31;
            float4 f = *(const float4*)(W + (size_t)(n0 + n) * Dm + kt + c);
            Bs[(c + 0) * 32 + n] = (bf16_t)f.x;
            Bs[(c + 1) * 32 + n] = (bf16_t)f.y;
            Bs[(c + 2) * 32 + n] = (bf16_t)f.z;
            Bs[(c + 3) * 32 + n] = (bf16_t)f.w;
        }
        async_wait();
        __syncthreads();
        v16bf a0 = frag_a(As, wv * 16, 32, 0, lane);
        v16bf a1 = frag_a(As, 64 + wv * 16, 32, 0, lane);
        #pragma unroll
        for (int f = 0; f < 2; ++f) {
            v16bf bb = frag_b(Bs, 0, 32, f * 16, lane);
            acc[0][f] = wmma_bf16(a0, bb, acc[0][f]);
            acc[1][f] = wmma_bf16(a1, bb, acc[1][f]);
        }
        __syncthreads();
    }

    const int mloc = (lane & 16) ? 8 : 0;
    const int ncol = lane & 15;
    #pragma unroll
    for (int st = 0; st < 2; ++st)
        #pragma unroll
        for (int f = 0; f < 2; ++f) {
            int n = n0 + f * 16 + ncol;
            float bn = bias[n];
            #pragma unroll
            for (int r = 0; r < 8; ++r) {
                int mg = m0 + st * 64 + wv * 16 + mloc + r;
                Out[(size_t)mg * Dm + n] = acc[st][f][r] + bn;
            }
        }
}

// ---------------------------------------------------------------------------
extern "C" void kernel_launch(void* const* d_in, const int* in_sizes, int n_in,
                              void* d_out, int out_size, void* d_ws, size_t ws_size,
                              hipStream_t stream) {
    const float* q      = (const float*)d_in[0];
    const float* k      = (const float*)d_in[1];
    const float* v      = (const float*)d_in[2];
    const float* cosT   = (const float*)d_in[3];
    const float* sinT   = (const float*)d_in[4];
    const float* mask   = (const float*)d_in[5];
    const float* shortM = (const float*)d_in[6];
    const float* Wq = (const float*)d_in[7];  const float* bq = (const float*)d_in[8];
    const float* Wk = (const float*)d_in[9];  const float* bk = (const float*)d_in[10];
    const float* Wv = (const float*)d_in[11]; const float* bv = (const float*)d_in[12];
    const float* Wo = (const float*)d_in[13]; const float* bo = (const float*)d_in[14];
    const float* pb = (const float*)d_in[15];

    // workspace: 4 bf16 tensors of B*S*D elements (64 MB total, L2-resident)
    const size_t TN = (size_t)Bm * Sm * Dm;
    bf16_t* qh     = (bf16_t*)d_ws;
    bf16_t* kh     = qh + TN;
    bf16_t* vh     = kh + TN;
    bf16_t* concat = vh + TN;

    dim3 blk(128);
    dim3 gProj(Bm * Sm / 128, (Dm / 2) / 32);          // 64 x 16
    proj_rope<<<gProj, blk, 0, stream>>>(q, Wq, bq, cosT, sinT, qh, 1);
    proj_rope<<<gProj, blk, 0, stream>>>(k, Wk, bk, cosT, sinT, kh, 1);
    proj_rope<<<gProj, blk, 0, stream>>>(v, Wv, bv, cosT, sinT, vh, 0);

    flash_attn<<<dim3(Sm / 64, Hn, Bm), blk, 0, stream>>>(qh, kh, vh, mask,
                                                          shortM, pb, concat);

    out_proj<<<dim3(Bm * Sm / 128, Dm / 32), blk, 0, stream>>>(concat, Wo, bo,
                                                               (float*)d_out);
}
